// R_CLCM_72164040507480
// MI455X (gfx1250) — matrix operand order repaired and checked by reading, not compiled, not run on hardware
//
#include <hip/hip_runtime.h>
#include <hip/hip_bf16.h>

typedef __attribute__((ext_vector_type(16))) _Float16 v16h;
typedef __attribute__((ext_vector_type(8)))  float    v8f;

#define EPSBN 1e-5f
#define NEG_INF -3.0e38f

// ---------------------------------------------------------------------------
// Channel attention: per-(b,c) mean & max over H*W
// ---------------------------------------------------------------------------
__global__ void ca_reduce_kernel(const float* __restrict__ x, int HW,
                                 float* __restrict__ avg, float* __restrict__ mx) {
  int bc = blockIdx.x;
  const float* p = x + (size_t)bc * HW;
  float s = 0.f, m = NEG_INF;
  for (int i = threadIdx.x; i < HW; i += blockDim.x) {
    float v = p[i];
    s += v;
    m = fmaxf(m, v);
  }
  __shared__ float ls[256];
  __shared__ float lm[256];
  ls[threadIdx.x] = s;
  lm[threadIdx.x] = m;
  __syncthreads();
  for (int off = 128; off > 0; off >>= 1) {
    if ((int)threadIdx.x < off) {
      ls[threadIdx.x] += ls[threadIdx.x + off];
      lm[threadIdx.x] = fmaxf(lm[threadIdx.x], lm[threadIdx.x + off]);
    }
    __syncthreads();
  }
  if (threadIdx.x == 0) {
    avg[bc] = ls[0] / (float)HW;
    mx[bc]  = lm[0];
  }
}

// Shared-MLP over avg & max descriptors -> sigmoid mask per (b,c)
__global__ void ca_mlp_kernel(const float* __restrict__ avg, const float* __restrict__ mx,
                              const float* __restrict__ w1, const float* __restrict__ b1,
                              const float* __restrict__ w2, const float* __restrict__ b2,
                              int C, float* __restrict__ mask) {
  int b = blockIdx.x;
  int Ch = C >> 1;                 // hidden width (<=256)
  __shared__ float za[256];
  __shared__ float zm[256];
  const float* av = avg + (size_t)b * C;
  const float* mv = mx  + (size_t)b * C;
  for (int h = threadIdx.x; h < Ch; h += blockDim.x) {
    float sa = b1[h], sm = b1[h];
    const float* wr = w1 + (size_t)h * C;
    for (int c = 0; c < C; ++c) { sa += wr[c] * av[c]; sm += wr[c] * mv[c]; }
    za[h] = fmaxf(sa, 0.f);
    zm[h] = fmaxf(sm, 0.f);
  }
  __syncthreads();
  for (int c = threadIdx.x; c < C; c += blockDim.x) {
    float sa = b2[c], sm = b2[c];
    const float* wr = w2 + (size_t)c * Ch;
    for (int h = 0; h < Ch; ++h) { sa += wr[h] * za[h]; sm += wr[h] * zm[h]; }
    float v = sa + sm;
    mask[(size_t)b * C + c] = 1.f / (1.f + __expf(-v));
  }
}

__global__ void ca_apply_kernel(const float* __restrict__ x, const float* __restrict__ mask,
                                int HW, size_t total, float* __restrict__ y) {
  size_t i = (size_t)blockIdx.x * blockDim.x + threadIdx.x;
  if (i >= total) return;
  size_t bc = i / (size_t)HW;
  y[i] = x[i] * mask[bc];
}

// ---------------------------------------------------------------------------
// Implicit-GEMM conv via V_WMMA_F32_16X16X32_F16, block-cooperative.
//
// Block = G waves (G = min(O/16, 8)); block covers one (batch, 32-wide spatial
// strip) x G oc-tiles. Per 32-deep K chunk, the im2col B tile (32K x 32N) is
// gathered ONCE into LDS by the whole block; every wave then reads its two f16
// B fragments from LDS and runs two WMMAs sharing one A fragment (N=32 per
// wave, two accumulators). Weights are streamed with global_prefetch.
//
// mode 0: out = acc + bias
// mode 1: out = relu(bn(acc + bias))
// mode 2: out = gamma * relu(bn(acc + bias)) + res
// ---------------------------------------------------------------------------
template<int KS>
__global__ __launch_bounds__(256)
void conv_wmma_kernel(const float* __restrict__ x, int B, int Ci, int H, int W,
                      const float* __restrict__ wgt, int pad, int O,
                      const float* __restrict__ bias,
                      const float* __restrict__ bng, const float* __restrict__ bnb,
                      const float* __restrict__ bnm, const float* __restrict__ bnv,
                      int mode, const float* __restrict__ gamma,
                      const float* __restrict__ res,
                      float* __restrict__ out) {
  const int HW = H * W;
  const int Ktot = Ci * KS * KS;
  const int G = blockDim.x >> 5;          // waves per block
  const int ocGroups = (O >> 4) / G;
  const int nPairs = HW >> 5;

  int blk   = blockIdx.x;
  int ocGrp = blk % ocGroups;
  int t2    = blk / ocGroups;
  int nP    = t2 % nPairs;
  int b     = t2 / nPairs;

  int waveId = threadIdx.x >> 5;
  int lane   = threadIdx.x & 31;
  int hi     = lane >> 4;
  int l15    = lane & 15;
  int ocT    = ocGrp * G + waveId;

  const int nBase = nP * 32;
  const float* wrow = wgt + (size_t)(ocT * 16 + l15) * Ktot;  // A row m = l15
  const int kbA = hi ? 8 : 0;   // A: lanes16-31 hold K+8 (v0-3) / K+24 (v4-7)
  const int kbB = hi ? 8 : 0;   // B: lanes16-31 hold K+8 (j0-7) / K+24 (j8-15)

  __shared__ float bt[32][33];  // padded: conflict-free column reads

  v8f acc0 = {}, acc1 = {};
  for (int k0 = 0; k0 < Ktot; k0 += 32) {
    // ---- block-cooperative im2col gather of B chunk (32K x 32N) ----
    __syncthreads();
    for (int t = threadIdx.x; t < 32 * 32; t += blockDim.x) {
      int kl = t >> 5, nn = t & 31;
      int kk = k0 + kl;
      float xv = 0.f;
      if (kk < Ktot) {
        int n  = nBase + nn;
        int oh = n / W, ow = n - oh * W;
        int ci  = kk / (KS * KS);
        int rem = kk - ci * (KS * KS);
        int kh  = rem / KS, kw = rem - kh * KS;
        int ih  = oh + kh - pad, iw = ow + kw - pad;
        if (ih >= 0 && ih < H && iw >= 0 && iw < W)
          xv = x[(((size_t)b * Ci + ci) * H + ih) * W + iw];
      }
      bt[kl][nn] = xv;
    }
    __syncthreads();

    // ---- A fragment (16x32 f16, ISA layout), streamed from global ----
    v16h a;
#pragma unroll
    for (int j = 0; j < 16; ++j) {
      int v  = j >> 1, lo = j & 1;
      int kk = (v < 4) ? (k0 + kbA + 2 * v + lo)
                       : (k0 + 16 + kbA + 2 * (v - 4) + lo);
      a[j] = (_Float16)((kk < Ktot) ? wrow[kk] : 0.f);
    }
    if (k0 + 32 < Ktot) __builtin_prefetch(wrow + k0 + 32, 0, 1);

    // ---- B fragments (32x16 f16 each) from LDS ----
    v16h b0, b1;
#pragma unroll
    for (int j = 0; j < 16; ++j) {
      b0[j] = (_Float16)bt[kbB + j + ((j & 8) ? 8 : 0)][l15];
      b1[j] = (_Float16)bt[kbB + j + ((j & 8) ? 8 : 0)][16 + l15];
    }
    acc0 = __builtin_amdgcn_wmma_f32_16x16x32_f16(false, a, false, b0,
                                                  (short)0, acc0, false, false);
    acc1 = __builtin_amdgcn_wmma_f32_16x16x32_f16(false, a, false, b1,
                                                  (short)0, acc1, false, false);
  }

  // ---- epilogue: D layout VGPR r -> rows r (lo half) / r+8 (hi half) ----
  int n0 = nBase + l15, n1 = n0 + 16;
#pragma unroll
  for (int r = 0; r < 8; ++r) {
    int oc = ocT * 16 + r + (hi ? 8 : 0);
    size_t row = ((size_t)b * O + oc) * HW;
    float v0 = acc0[r], v1 = acc1[r];
    if (mode == 0) {
      float bi = bias[oc];
      v0 += bi; v1 += bi;
    } else {
      float s  = bng[oc] * rsqrtf(bnv[oc] + EPSBN);
      float sh = bias[oc] * s + bnb[oc] - bnm[oc] * s;
      v0 = fmaxf(v0 * s + sh, 0.f);
      v1 = fmaxf(v1 * s + sh, 0.f);
      if (mode == 2) {
        float g = gamma[0];
        v0 = g * v0 + res[row + n0];
        v1 = g * v1 + res[row + n1];
      }
    }
    out[row + n0] = v0;
    out[row + n1] = v1;
  }
}

// ---------------------------------------------------------------------------
// Prototype soft-assignment on r4 (B,64,8,8) with protos (20,64)
// ---------------------------------------------------------------------------
__global__ void prototype_kernel(const float* __restrict__ x, const float* __restrict__ protos,
                                 int B, float* __restrict__ y) {
  int pix = blockIdx.x * blockDim.x + threadIdx.x;
  if (pix >= B * 64) return;
  int b = pix >> 6, n = pix & 63;
  float f[64];
#pragma unroll
  for (int c = 0; c < 64; ++c) f[c] = x[((size_t)b * 64 + c) * 64 + n];
  float lg[20];
  float mxv = NEG_INF;
  for (int k = 0; k < 20; ++k) {
    float s = 0.f;
    for (int c = 0; c < 64; ++c) s += f[c] * protos[k * 64 + c];
    lg[k] = s;
    mxv = fmaxf(mxv, s);
  }
  float den = 0.f;
  for (int k = 0; k < 20; ++k) { lg[k] = __expf(lg[k] - mxv); den += lg[k]; }
  float inv = 1.f / den;
  for (int c = 0; c < 64; ++c) {
    float rec = 0.f;
    for (int k = 0; k < 20; ++k) rec += lg[k] * protos[k * 64 + c];
    y[((size_t)b * 64 + c) * 64 + n] = f[c] + rec * inv;
  }
}

// ---------------------------------------------------------------------------
// Non-local attention, streamed (no materialized score matrix).
// Pass 1: per row m, online softmax stats (max, 1/sum) over all n.
// Pass 2: per column n, rebuild s[m,n], apply weights, out = k + sum_m w*q[:,m].
// q: (B,16,Nm)  k: (B,16,Nn)  stats: (B,Nm,2)  out: (B,16,Nn)
// ---------------------------------------------------------------------------
__global__ void att_stats_kernel(const float* __restrict__ q, const float* __restrict__ km,
                                 int Nm, int Nn, float* __restrict__ stats) {
  int b = blockIdx.y;
  int m = blockIdx.x * blockDim.x + threadIdx.x;
  bool act = (m < Nm);
  float qv[16];
  if (act) {
#pragma unroll
    for (int c = 0; c < 16; ++c) qv[c] = q[((size_t)b * 16 + c) * Nm + m];
  }
  __shared__ float lk[16][64];
  float rmax = NEG_INF, rsum = 0.f;
  for (int n0 = 0; n0 < Nn; n0 += 64) {
    __syncthreads();
    for (int t = threadIdx.x; t < 16 * 64; t += blockDim.x) {
      int c = t >> 6, j = t & 63;
      lk[c][j] = km[((size_t)b * 16 + c) * Nn + n0 + j];
    }
    __syncthreads();
    if (act) {
      for (int j = 0; j < 64; ++j) {
        float s = 0.f;
#pragma unroll
        for (int c = 0; c < 16; ++c) s += qv[c] * lk[c][j];
        float nm = fmaxf(rmax, s);
        rsum = rsum * __expf(rmax - nm) + __expf(s - nm);
        rmax = nm;
      }
    }
  }
  if (act) {
    stats[((size_t)b * Nm + m) * 2 + 0] = rmax;
    stats[((size_t)b * Nm + m) * 2 + 1] = 1.f / rsum;
  }
}

__global__ void att_apply_kernel(const float* __restrict__ q, const float* __restrict__ km,
                                 int Nm, int Nn, const float* __restrict__ stats,
                                 float* __restrict__ out) {
  int b = blockIdx.y;
  int n = blockIdx.x * blockDim.x + threadIdx.x;   // Nn is a multiple of 256
  float kv[16], acc[16];
#pragma unroll
  for (int c = 0; c < 16; ++c) {
    kv[c] = km[((size_t)b * 16 + c) * Nn + n];
    acc[c] = 0.f;
  }
  __shared__ float lq[16][64];
  __shared__ float lst[64][2];
  for (int m0 = 0; m0 < Nm; m0 += 64) {
    __syncthreads();
    for (int t = threadIdx.x; t < 16 * 64; t += blockDim.x) {
      int c = t >> 6, j = t & 63;
      lq[c][j] = (m0 + j < Nm) ? q[((size_t)b * 16 + c) * Nm + m0 + j] : 0.f;
    }
    if (threadIdx.x < 64) {
      int mm = m0 + threadIdx.x;
      lst[threadIdx.x][0] = (mm < Nm) ? stats[((size_t)b * Nm + mm) * 2 + 0] : 0.f;
      lst[threadIdx.x][1] = (mm < Nm) ? stats[((size_t)b * Nm + mm) * 2 + 1] : 0.f;
    }
    __syncthreads();
    int lim = (Nm - m0 < 64) ? (Nm - m0) : 64;
    for (int j = 0; j < lim; ++j) {
      float s = 0.f;
#pragma unroll
      for (int c = 0; c < 16; ++c) s += kv[c] * lq[c][j];
      float w = __expf(s - lst[j][0]) * lst[j][1];
#pragma unroll
      for (int c = 0; c < 16; ++c) acc[c] += w * lq[c][j];
    }
  }
#pragma unroll
  for (int c = 0; c < 16; ++c)
    out[((size_t)b * 16 + c) * Nn + n] = kv[c] + acc[c];
}

// ---------------------------------------------------------------------------
// Host side
// ---------------------------------------------------------------------------
namespace {

struct Params {
  const float *x[4];
  const float *ca_w1[4], *ca_b1[4], *ca_w2[4], *ca_b2[4];
  // 0-3: conv_r1..r4 ; 4-6: c1r..c3r
  const float *cw[7], *cbias[7], *cg[7], *cb[7], *cm[7], *cv[7];
  const float *nw[4], *nb[4];
  const float *ow[3], *ob[3];
  const float *gam[3];
  const float *protos;
};

// layout A: JAX pytree flatten (dict keys sorted at every level)
static const int kSortedSizes[80] = {
  64,64,64,64,64,1024,        // c1r: bias, bn.b, bn.g, bn.m, bn.v, w
  64,64,64,64,64,1024,        // c2r
  64,64,64,64,64,1024,        // c3r
  32,64,2048,2048,            // ca1: b1,b2,w1,w2
  64,128,8192,8192,           // ca2
  128,256,32768,32768,        // ca3
  256,512,131072,131072,      // ca4
  64,64,64,64,64,4096,        // conv_r1
  64,64,64,64,64,73728,       // conv_r2
  64,64,64,64,64,147456,      // conv_r3
  64,64,64,64,64,294912,      // conv_r4
  1,1,1,                      // gam1..3
  16,1024,16,1024,16,1024,16,1024, // n1b,n1w .. n4b,n4w
  64,4096,128,73728,256,147456,    // o1b,o1w,o2b,o2w,o3b,o3w
  1280,                       // protos
  4194304,2097152,1048576,524288   // x1..x4
};

// layout B: python dict insertion order
static const int kInsertSizes[80] = {
  4194304,2097152,1048576,524288,  // x1..x4
  2048,32,2048,64,                 // ca1: w1,b1,w2,b2
  8192,64,8192,128,
  32768,128,32768,256,
  131072,256,131072,512,
  4096,64,64,64,64,64,             // conv_r1: w,bias,g,b,m,v
  73728,64,64,64,64,64,
  147456,64,64,64,64,64,
  294912,64,64,64,64,64,
  1024,16,1024,16,1024,16,1024,16, // n1w,n1b .. n4w,n4b
  1024,64,64,64,64,64,             // c1r
  1024,64,64,64,64,64,
  1024,64,64,64,64,64,
  4096,64,73728,128,147456,256,    // o1w,o1b .. o3w,o3b
  1,1,1,
  1280
};

static bool match(const int* s, int n_in, const int* e) {
  if (n_in != 80) return false;
  for (int i = 0; i < 80; ++i) if (s[i] != e[i]) return false;
  return true;
}

static void fill_sorted(void* const* d, Params& p) {
  auto F = [&](int i) { return (const float*)d[i]; };
  for (int i = 0; i < 3; ++i) {  // c1r,c2r,c3r -> slots 4..6
    int base = i * 6;
    p.cbias[4 + i] = F(base + 0); p.cb[4 + i] = F(base + 1);
    p.cg[4 + i] = F(base + 2);    p.cm[4 + i] = F(base + 3);
    p.cv[4 + i] = F(base + 4);    p.cw[4 + i] = F(base + 5);
  }
  for (int i = 0; i < 4; ++i) {  // ca1..ca4
    int base = 18 + i * 4;
    p.ca_b1[i] = F(base + 0); p.ca_b2[i] = F(base + 1);
    p.ca_w1[i] = F(base + 2); p.ca_w2[i] = F(base + 3);
  }
  for (int i = 0; i < 4; ++i) {  // conv_r1..r4 -> slots 0..3
    int base = 34 + i * 6;
    p.cbias[i] = F(base + 0); p.cb[i] = F(base + 1);
    p.cg[i] = F(base + 2);    p.cm[i] = F(base + 3);
    p.cv[i] = F(base + 4);    p.cw[i] = F(base + 5);
  }
  p.gam[0] = F(58); p.gam[1] = F(59); p.gam[2] = F(60);
  for (int i = 0; i < 4; ++i) { p.nb[i] = F(61 + 2 * i); p.nw[i] = F(62 + 2 * i); }
  for (int i = 0; i < 3; ++i) { p.ob[i] = F(69 + 2 * i); p.ow[i] = F(70 + 2 * i); }
  p.protos = F(75);
  for (int i = 0; i < 4; ++i) p.x[i] = F(76 + i);
}

static void fill_insert(void* const* d, Params& p) {
  auto F = [&](int i) { return (const float*)d[i]; };
  for (int i = 0; i < 4; ++i) p.x[i] = F(i);
  for (int i = 0; i < 4; ++i) {
    int base = 4 + i * 4;
    p.ca_w1[i] = F(base + 0); p.ca_b1[i] = F(base + 1);
    p.ca_w2[i] = F(base + 2); p.ca_b2[i] = F(base + 3);
  }
  for (int i = 0; i < 4; ++i) {  // conv_r*: w,bias,g,b,m,v
    int base = 20 + i * 6;
    p.cw[i] = F(base + 0); p.cbias[i] = F(base + 1);
    p.cg[i] = F(base + 2); p.cb[i] = F(base + 3);
    p.cm[i] = F(base + 4); p.cv[i] = F(base + 5);
  }
  for (int i = 0; i < 4; ++i) { p.nw[i] = F(44 + 2 * i); p.nb[i] = F(45 + 2 * i); }
  for (int i = 0; i < 3; ++i) {  // c1r..c3r
    int base = 52 + i * 6;
    p.cw[4 + i] = F(base + 0); p.cbias[4 + i] = F(base + 1);
    p.cg[4 + i] = F(base + 2); p.cb[4 + i] = F(base + 3);
    p.cm[4 + i] = F(base + 4); p.cv[4 + i] = F(base + 5);
  }
  for (int i = 0; i < 3; ++i) { p.ow[i] = F(70 + 2 * i); p.ob[i] = F(71 + 2 * i); }
  p.gam[0] = F(76); p.gam[1] = F(77); p.gam[2] = F(78);
  p.protos = F(79);
}

} // namespace

extern "C" void kernel_launch(void* const* d_in, const int* in_sizes, int n_in,
                              void* d_out, int out_size, void* d_ws, size_t ws_size,
                              hipStream_t stream) {
  Params p;
  if (match(in_sizes, n_in, kSortedSizes))      fill_sorted(d_in, p);
  else if (match(in_sizes, n_in, kInsertSizes)) fill_insert(d_in, p);
  else if (n_in == 80 && in_sizes[0] > 1000000) fill_insert(d_in, p);
  else                                          fill_sorted(d_in, p);

  const int B = 16;
  // ---- workspace carve-up (floats) ----
  float* ws = (float*)d_ws;
  float* Y1  = ws;               // ca(x1): 16*64*64*64
  float* Y2  = Y1  + 4194304;    // ca(x2): 16*128*32*32
  float* Y3  = Y2  + 2097152;    // ca(x3)
  float* Y4  = Y3  + 1048576;    // ca(x4)
  float* R1  = Y4  + 524288;     // (16,64,64,64)
  float* R2  = R1  + 4194304;    // (16,64,32,32)
  float* R3  = R2  + 1048576;    // (16,64,16,16)
  float* R4  = R3  + 262144;     // (16,64,8,8)
  float* R4P = R4  + 65536;      // prototype(r4)
  float* Q4  = R4P + 65536;      // (16,16,64)
  float* Q3  = Q4  + 16384;      // (16,16,256)
  float* Q2  = Q3  + 65536;      // (16,16,1024)
  float* Q1  = Q2  + 262144;     // (16,16,4096)
  float* Q3I = Q1  + 1048576;    // r_3_in
  float* Q2I = Q3I + 65536;      // r_2_in
  float* Q1I = Q2I + 262144;     // r_1_in
  float* ST  = Q1I + 1048576;    // attention row stats (2*16*1024 max)
  float* OR3 = ST  + 32768;      // out_r3
  float* OR2 = OR3 + 262144;     // out_r2
  float* OR1 = OR2 + 1048576;    // out_r1
  float* AVG = OR1 + 4194304;    // descriptors (max B*C = 8192 each)
  float* MAXD = AVG + 8192;
  float* MASK = MAXD + 8192;
  size_t need = (size_t)(MASK + 8192 - ws) * sizeof(float);
  if (ws_size < need) return;

  float* out = (float*)d_out;
  float* O1 = out;               // (16,64,64,64)
  float* O2 = out + 4194304;     // (16,128,32,32)
  float* O3 = out + 6291456;     // (16,256,16,16)

  auto conv = [&](const float* xin, int Ci, int H, int W, const float* wgt, int ks, int pad,
                  int O, const float* bias, const float* g, const float* bb,
                  const float* m, const float* v, int mode, const float* gamma,
                  const float* res, float* yout) {
    int HW = H * W;
    int ocTiles = O / 16;
    int G = ocTiles < 8 ? ocTiles : 8;      // waves per block
    int ocGroups = ocTiles / G;
    int nPairs = HW / 32;
    int blocks = B * nPairs * ocGroups;
    dim3 blockDim3(32 * G);
    if (ks == 1)
      conv_wmma_kernel<1><<<blocks, blockDim3, 0, stream>>>(xin, B, Ci, H, W, wgt, pad, O,
                                                            bias, g, bb, m, v, mode, gamma, res, yout);
    else
      conv_wmma_kernel<3><<<blocks, blockDim3, 0, stream>>>(xin, B, Ci, H, W, wgt, pad, O,
                                                            bias, g, bb, m, v, mode, gamma, res, yout);
  };

  auto att = [&](const float* q, const float* km, int Nm, int Nn, float* yout) {
    dim3 g1((Nm + 255) / 256, B);
    att_stats_kernel<<<g1, 256, 0, stream>>>(q, km, Nm, Nn, ST);
    dim3 g2(Nn / 256, B);
    att_apply_kernel<<<g2, 256, 0, stream>>>(q, km, Nm, Nn, ST, yout);
  };

  // ---- Stage 1: channel attention on each level ----
  const int   Cs[4]  = {64, 128, 256, 512};
  const int   Hs[4]  = {64, 32, 16, 8};
  float*      Ys[4]  = {Y1, Y2, Y3, Y4};
  for (int i = 0; i < 4; ++i) {
    int C = Cs[i], HW = Hs[i] * Hs[i];
    size_t total = (size_t)B * C * HW;
    ca_reduce_kernel<<<B * C, 256, 0, stream>>>(p.x[i], HW, AVG, MAXD);
    ca_mlp_kernel<<<B, 256, 0, stream>>>(AVG, MAXD, p.ca_w1[i], p.ca_b1[i],
                                         p.ca_w2[i], p.ca_b2[i], C, MASK);
    ca_apply_kernel<<<(int)((total + 255) / 256), 256, 0, stream>>>(p.x[i], MASK, HW, total, Ys[i]);
  }

  // ---- Stage 2: conv blocks r1..r4 (bias+BN+ReLU) ----
  conv(Y1, 64,  64, 64, p.cw[0], 1, 0, 64, p.cbias[0], p.cg[0], p.cb[0], p.cm[0], p.cv[0], 1, nullptr, nullptr, R1);
  conv(Y2, 128, 32, 32, p.cw[1], 3, 1, 64, p.cbias[1], p.cg[1], p.cb[1], p.cm[1], p.cv[1], 1, nullptr, nullptr, R2);
  conv(Y3, 256, 16, 16, p.cw[2], 3, 1, 64, p.cbias[2], p.cg[2], p.cb[2], p.cm[2], p.cv[2], 1, nullptr, nullptr, R3);
  conv(Y4, 512, 8,  8,  p.cw[3], 3, 1, 64, p.cbias[3], p.cg[3], p.cb[3], p.cm[3], p.cv[3], 1, nullptr, nullptr, R4);

  // ---- Stage 3: prototype residual on r4 ----
  prototype_kernel<<<4, 256, 0, stream>>>(R4, p.protos, B, R4P);

  // ---- Stage 4: 1x1 projections to 16 channels ----
  conv(R4P, 64, 8,  8,  p.nw[3], 1, 0, 16, p.nb[3], nullptr, nullptr, nullptr, nullptr, 0, nullptr, nullptr, Q4);
  conv(R3,  64, 16, 16, p.nw[2], 1, 0, 16, p.nb[2], nullptr, nullptr, nullptr, nullptr, 0, nullptr, nullptr, Q3);
  conv(R2,  64, 32, 32, p.nw[1], 1, 0, 16, p.nb[1], nullptr, nullptr, nullptr, nullptr, 0, nullptr, nullptr, Q2);
  conv(R1,  64, 64, 64, p.nw[0], 1, 0, 16, p.nb[0], nullptr, nullptr, nullptr, nullptr, 0, nullptr, nullptr, Q1);

  // ---- Stage 5: cross-level non-local chain ----
  att(Q4,  Q3, 64,   256,  Q3I);
  att(Q3I, Q2, 256,  1024, Q2I);
  att(Q2I, Q1, 1024, 4096, Q1I);

  // ---- Stage 6: gamma * convblock + residual ----
  conv(Q3I, 16, 16, 16, p.cw[6], 1, 0, 64, p.cbias[6], p.cg[6], p.cb[6], p.cm[6], p.cv[6], 2, p.gam[2], R3, OR3);
  conv(Q2I, 16, 32, 32, p.cw[5], 1, 0, 64, p.cbias[5], p.cg[5], p.cb[5], p.cm[5], p.cv[5], 2, p.gam[1], R2, OR2);
  conv(Q1I, 16, 64, 64, p.cw[4], 1, 0, 64, p.cbias[4], p.cg[4], p.cb[4], p.cm[4], p.cv[4], 2, p.gam[0], R1, OR1);

  // ---- Stage 7: output convs ----
  conv(OR1, 64, 64, 64, p.ow[0], 1, 0, 64,  p.ob[0], nullptr, nullptr, nullptr, nullptr, 0, nullptr, nullptr, O1);
  conv(OR2, 64, 32, 32, p.ow[1], 3, 1, 128, p.ob[1], nullptr, nullptr, nullptr, nullptr, 0, nullptr, nullptr, O2);
  conv(OR3, 64, 16, 16, p.ow[2], 3, 1, 256, p.ob[2], nullptr, nullptr, nullptr, nullptr, 0, nullptr, nullptr, O3);
}
